// LSTMCRF_8787503088099
// MI455X (gfx1250) — matrix operand order repaired and checked by reading, not compiled, not run on hardware
//
#include <hip/hip_runtime.h>
#include <hip/hip_bf16.h>
#include <math.h>

// ---------------------------------------------------------------------------
// Problem constants (from reference)
// ---------------------------------------------------------------------------
#define VSZ 50000
#define E   128
#define H   256
#define G3  768          // 3*H
#define KTAG 9
#define BATCH 64
#define TLEN 512
#define MROWS (BATCH*TLEN)   // 32768

typedef __attribute__((ext_vector_type(16))) __bf16 v16bf;
typedef __attribute__((ext_vector_type(8)))  float  v8f;
typedef int v4i __attribute__((vector_size(16)));

__device__ __forceinline__ unsigned short f2bf_bits(float f) {
    union { float f; unsigned u; } v; v.f = f;
    unsigned r = v.u + 0x7FFFu + ((v.u >> 16) & 1u);   // round-to-nearest-even
    return (unsigned short)(r >> 16);
}

__device__ __forceinline__ v8f vzero8() {
    v8f v;
    #pragma unroll
    for (int i = 0; i < 8; ++i) v[i] = 0.0f;
    return v;
}

__device__ __forceinline__ void wait_async0() {
#if __has_builtin(__builtin_amdgcn_s_wait_asynccnt)
    __builtin_amdgcn_s_wait_asynccnt(0);
#else
    asm volatile("s_wait_asynccnt 0" ::: "memory");
#endif
}

// ---------------------------------------------------------------------------
// Kernel 0: convert an (N x K) fp32 row-major weight into bf16 WMMA B-fragment
// major layout.  Fragment f = nt*KT + kt holds a 32x16 (KxN) tile; within the
// fragment, element r = lane*16 + il with lane = n%16 + 16*((k%32>>3)&1),
// il = (k&7) + 8*(k%32>=16)  (ISA 7.12.2 16-bit B layout, wave32).
// ---------------------------------------------------------------------------
__global__ void conv_frag_kernel(const float* __restrict__ w,
                                 unsigned short* __restrict__ dst,
                                 int N, int K) {
    int KT = K >> 5;
    int total = N * K;
    for (int o = blockIdx.x * blockDim.x + threadIdx.x; o < total;
         o += gridDim.x * blockDim.x) {
        int f = o >> 9, r = o & 511;
        int lane = r >> 4, il = r & 15;
        int nn = lane & 15;
        int kk = (il & 7) + ((lane >> 4) << 3) + ((il >> 3) << 4);
        int nt = f / KT, kt = f - nt * KT;
        int n = nt * 16 + nn;
        int k = kt * 32 + kk;
        dst[o] = f2bf_bits(w[(size_t)n * K + k]);
    }
}

// ---------------------------------------------------------------------------
// Kernel 1: embedding gather + xg = embed @ w_ih^T + b_ih   (both directions)
// grid.x = M/16 tiles (2048), grid.y = direction; block = 256 (8 waves).
// A tile (16x128 bf16) staged in LDS in A-fragment layout; B fragments come
// pre-swizzled from workspace.  Output xg is written TIME-MAJOR (T, B, 3H) so
// the serial GRU scan reads one contiguous 196KB slice per step.
// ---------------------------------------------------------------------------
__global__ __launch_bounds__(256) void embed_xg_kernel(
    const int* __restrict__ x, const float* __restrict__ etab,
    const unsigned short* __restrict__ wih_f, const unsigned short* __restrict__ wih_b,
    const float* __restrict__ bih_f, const float* __restrict__ bih_b,
    float* __restrict__ xg_f, float* __restrict__ xg_b) {
    __shared__ __align__(64) unsigned short sA[4 * 512];   // 4 k-tiles
    const int mtile = blockIdx.x;
    const int dir   = blockIdx.y;
    const unsigned short* wih = dir ? wih_b : wih_f;
    const float* bih = dir ? bih_b : bih_f;
    float* xg = dir ? xg_b : xg_f;

    const int tid = threadIdx.x, lane = tid & 31, wave = tid >> 5;

    // stage 16x128 embedding tile -> bf16 A-fragment layout in LDS
    #pragma unroll
    for (int j = 0; j < 8; ++j) {
        int i = tid * 8 + j;                 // 2048 elements total
        int m = i >> 7, k = i & 127;
        int tok = x[mtile * 16 + m];
        float v = etab[(size_t)tok * E + k];
        int kt = k >> 5, kk = k & 31;
        int lane2 = m + (((kk >> 3) & 1) << 4);       // A layout (ISA 7.12.2)
        int il = (kk & 7) + ((kk >> 4) << 3);
        sA[kt * 512 + lane2 * 16 + il] = f2bf_bits(v);
    }
    __syncthreads();

    for (int nt = wave; nt < 48; nt += 8) {
        v8f acc = vzero8();
        #pragma unroll
        for (int kt = 0; kt < 4; ++kt) {
            v16bf a = *(const v16bf*)&sA[kt * 512 + lane * 16];
            const v16bf* bp = (const v16bf*)((const __bf16*)wih +
                               ((size_t)(nt * 4 + kt)) * 512 + lane * 16);
            acc = __builtin_amdgcn_wmma_f32_16x16x32_bf16(
                false, a, false, *bp, (short)0, acc, false, false);
        }
        int n = nt * 16 + (lane & 15);
        float bias = bih[n];
        #pragma unroll
        for (int r = 0; r < 8; ++r) {
            int m = mtile * 16 + r + 8 * (lane >> 4);  // C layout (ISA 7.12.2)
            int b = m >> 9, t = m & (TLEN - 1);        // m = b*T + t
            xg[((size_t)t * BATCH + b) * G3 + n] = acc[r] + bias;
        }
    }
}

// ---------------------------------------------------------------------------
// Kernel 2: persistent GRU, one block per direction (grid=2, block=1024).
// 32 waves; 64 jobs (mtile 0..3  x  hidden 16-col group 0..15), 2 jobs/wave.
// Per step: hg = h @ w_hh^T via 8 K-steps of bf16 WMMA per gate tile,
// then fused gate math; h kept in registers (fp32) + LDS bf16 A-fragments.
// xg and hs are time-major: each step touches one contiguous slice.
// ---------------------------------------------------------------------------
__global__ __launch_bounds__(1024, 1) void gru_kernel(
    const float* __restrict__ xg_f, const float* __restrict__ xg_b,
    const unsigned short* __restrict__ whh_f, const unsigned short* __restrict__ whh_b,
    const float* __restrict__ bhh_f, const float* __restrict__ bhh_b,
    float* __restrict__ hf, float* __restrict__ hb) {
    __shared__ __align__(64) unsigned short sH[4 * 8 * 512];   // 32 KB bf16 A-frags

    const int dir = blockIdx.x;
    const float* xg = dir ? xg_b : xg_f;
    const unsigned short* whh = dir ? whh_b : whh_f;
    const float* bhh = dir ? bhh_b : bhh_f;
    float* hs = dir ? hb : hf;

    const int tid = threadIdx.x, lane = tid & 31, wave = tid >> 5;
    const int nhi = lane & 15, rowsel = lane >> 4;

    for (int i = tid; i < 4 * 8 * 512; i += 1024) sH[i] = 0;   // h0 = 0
    float hreg[2][8];
    #pragma unroll
    for (int j = 0; j < 2; ++j)
        #pragma unroll
        for (int r = 0; r < 8; ++r) hreg[j][r] = 0.0f;
    __syncthreads();

    for (int s = 0; s < TLEN; ++s) {
        const int t = dir ? (TLEN - 1 - s) : s;
        #pragma unroll
        for (int jj = 0; jj < 2; ++jj) {
            const int job = wave + jj * 32;
            const int mtile = job >> 4, ncol = job & 15;
            const int nh = ncol * 16 + nhi;

            v8f acc0 = vzero8(), acc1 = vzero8(), acc2 = vzero8();
            #pragma unroll
            for (int kt = 0; kt < 8; ++kt) {
                v16bf a = *(const v16bf*)&sH[(mtile * 8 + kt) * 512 + lane * 16];
                const __bf16* wb = (const __bf16*)whh;
                const v16bf* b0 = (const v16bf*)(wb + ((size_t)((0 * 16 + ncol) * 8 + kt)) * 512 + lane * 16);
                const v16bf* b1 = (const v16bf*)(wb + ((size_t)((1 * 16 + ncol) * 8 + kt)) * 512 + lane * 16);
                const v16bf* b2 = (const v16bf*)(wb + ((size_t)((2 * 16 + ncol) * 8 + kt)) * 512 + lane * 16);
                __builtin_prefetch(b2 + 1, 0, 0);    // global_prefetch_b8
                acc0 = __builtin_amdgcn_wmma_f32_16x16x32_bf16(false, a, false, *b0, (short)0, acc0, false, false);
                acc1 = __builtin_amdgcn_wmma_f32_16x16x32_bf16(false, a, false, *b1, (short)0, acc1, false, false);
                acc2 = __builtin_amdgcn_wmma_f32_16x16x32_bf16(false, a, false, *b2, (short)0, acc2, false, false);
            }
            const float bh0 = bhh[0 * H + nh], bh1 = bhh[1 * H + nh], bh2 = bhh[2 * H + nh];
            #pragma unroll
            for (int r = 0; r < 8; ++r) {
                int b = mtile * 16 + r + 8 * rowsel;
                size_t xrow = ((size_t)t * BATCH + b) * G3;   // time-major
                float xr = xg[xrow + 0 * H + nh];
                float xz = xg[xrow + 1 * H + nh];
                float xn = xg[xrow + 2 * H + nh];
                float rg = 1.0f / (1.0f + __expf(-(xr + acc0[r] + bh0)));
                float zg = 1.0f / (1.0f + __expf(-(xz + acc1[r] + bh1)));
                float ng = tanhf(xn + rg * (acc2[r] + bh2));
                float hn2 = (1.0f - zg) * ng + zg * hreg[jj][r];
                hreg[jj][r] = hn2;
                hs[((size_t)t * BATCH + b) * H + nh] = hn2;    // time-major
            }
        }
        __syncthreads();   // all reads of step-s fragments done
        #pragma unroll
        for (int jj = 0; jj < 2; ++jj) {
            const int job = wave + jj * 32;
            const int mtile = job >> 4, ncol = job & 15;
            const int nh = ncol * 16 + nhi;
            const int kt = nh >> 5, kk = nh & 31;
            const int il = (kk & 7) + ((kk >> 4) << 3);
            #pragma unroll
            for (int r = 0; r < 8; ++r) {
                int mlocal = r + 8 * rowsel;
                int lane2 = mlocal + (((kk >> 3) & 1) << 4);
                sH[(mtile * 8 + kt) * 512 + lane2 * 16 + il] = f2bf_bits(hreg[jj][r]);
            }
        }
        __syncthreads();   // fragments ready for step s+1
    }
}

// ---------------------------------------------------------------------------
// Kernel 3: logits = [hf|hb] @ w_lin^T + b_lin, softmax over K=9.
// One wave per (b,t) row.  w_lin (18 KB) is staged into LDS once per block
// via gfx1250 async global->LDS copies (ASYNCcnt path) when available.
// ---------------------------------------------------------------------------
__global__ __launch_bounds__(256) void linsoft_kernel(
    const float* __restrict__ hf, const float* __restrict__ hb,
    const float* __restrict__ w_lin, const float* __restrict__ b_lin,
    float* __restrict__ emis) {
    __shared__ __align__(16) float sW[KTAG * 2 * H];   // 4608 floats = 18 KB
    const int tid = threadIdx.x;
    const int lane = tid & 31;

#if __has_builtin(__builtin_amdgcn_global_load_async_to_lds_b128)
    // 4608 floats = 1152 x 16B async lane-copies (GLOBAL_LOAD_ASYNC_TO_LDS_B128)
    for (int i = tid; i < (KTAG * 2 * H) / 4; i += 256) {
        __builtin_amdgcn_global_load_async_to_lds_b128(
            (v4i __attribute__((address_space(1)))*)(w_lin + i * 4),
            (v4i __attribute__((address_space(3)))*)(sW + i * 4), 0, 0);
    }
    wait_async0();
#else
    for (int i = tid; i < KTAG * 2 * H; i += 256) sW[i] = w_lin[i];
#endif
    __syncthreads();

    const int row = blockIdx.x * 8 + (tid >> 5);
    const int b = row >> 9, t = row & (TLEN - 1);      // row = b*T + t
    const size_t hrow = (size_t)t * BATCH + b;         // time-major h

    float part[KTAG];
    #pragma unroll
    for (int k = 0; k < KTAG; ++k) part[k] = 0.0f;

    for (int d = lane; d < 2 * H; d += 32) {
        float v = (d < H) ? hf[hrow * H + d] : hb[hrow * H + (d - H)];
        #pragma unroll
        for (int k = 0; k < KTAG; ++k) part[k] += v * sW[k * (2 * H) + d];
    }
    #pragma unroll
    for (int k = 0; k < KTAG; ++k) {
        #pragma unroll
        for (int off = 16; off > 0; off >>= 1)
            part[k] += __shfl_xor(part[k], off, 32);
        part[k] += b_lin[k];
    }
    float mx = part[0];
    #pragma unroll
    for (int k = 1; k < KTAG; ++k) mx = fmaxf(mx, part[k]);
    float sum = 0.0f;
    #pragma unroll
    for (int k = 0; k < KTAG; ++k) sum += __expf(part[k] - mx);
    float inv = 1.0f / sum;
    if (lane < KTAG)
        emis[(size_t)row * KTAG + lane] = __expf(part[lane] - mx) * inv;
}

// ---------------------------------------------------------------------------
// Kernel 4: CRF forward algorithm + loss.  One thread per batch row (64).
// ---------------------------------------------------------------------------
__global__ __launch_bounds__(64) void crf_kernel(
    const int* __restrict__ x, const float* __restrict__ emis,
    const float* __restrict__ transitions, float* __restrict__ out) {
    __shared__ float tr[KTAG * KTAG];
    __shared__ float red[BATCH];
    const int b = threadIdx.x;
    if (b < KTAG * KTAG) tr[b] = transitions[b];
    __syncthreads();

    int len = 0;
    for (int t = 0; t < TLEN; ++t) len += (x[b * TLEN + t] >= 0) ? 1 : 0;

    float alpha[KTAG];
    #pragma unroll
    for (int k = 0; k < KTAG; ++k) alpha[k] = emis[((size_t)b * TLEN + 0) * KTAG + k];

    for (int t = 1; t < TLEN; ++t) {
        if (t >= len) continue;
        float na[KTAG];
        #pragma unroll
        for (int k = 0; k < KTAG; ++k) {
            float m = -3.4e38f;
            #pragma unroll
            for (int j = 0; j < KTAG; ++j) m = fmaxf(m, alpha[j] + tr[j * KTAG + k]);
            float s = 0.0f;
            #pragma unroll
            for (int j = 0; j < KTAG; ++j) s += __expf(alpha[j] + tr[j * KTAG + k] - m);
            na[k] = m + __logf(s) + emis[((size_t)b * TLEN + t) * KTAG + k];
        }
        #pragma unroll
        for (int k = 0; k < KTAG; ++k) alpha[k] = na[k];
    }
    float m = alpha[0];
    #pragma unroll
    for (int k = 1; k < KTAG; ++k) m = fmaxf(m, alpha[k]);
    float s = 0.0f;
    #pragma unroll
    for (int k = 0; k < KTAG; ++k) s += __expf(alpha[k] - m);
    red[b] = m + __logf(s);
    __syncthreads();
    if (b == 0) {
        float tot = 0.0f;
        for (int i = 0; i < BATCH; ++i) tot += red[i];
        out[0] = -tot;
    }
}

// ---------------------------------------------------------------------------
// Host-side launcher
// ---------------------------------------------------------------------------
extern "C" void kernel_launch(void* const* d_in, const int* in_sizes, int n_in,
                              void* d_out, int out_size, void* d_ws, size_t ws_size,
                              hipStream_t stream) {
    (void)in_sizes; (void)n_in; (void)out_size; (void)ws_size;
    const int*   x      = (const int*)  d_in[0];
    const float* etab   = (const float*)d_in[1];
    const float* w_ih_f = (const float*)d_in[2];
    const float* w_hh_f = (const float*)d_in[3];
    const float* b_ih_f = (const float*)d_in[4];
    const float* b_hh_f = (const float*)d_in[5];
    const float* w_ih_b = (const float*)d_in[6];
    const float* w_hh_b = (const float*)d_in[7];
    const float* b_ih_b = (const float*)d_in[8];
    const float* b_hh_b = (const float*)d_in[9];
    const float* w_lin  = (const float*)d_in[10];
    const float* b_lin  = (const float*)d_in[11];
    const float* trans  = (const float*)d_in[12];
    float* out = (float*)d_out;

    char* ws = (char*)d_ws;
    size_t off = 0;
    auto carve = [&](size_t bytes) -> char* {
        char* p = ws + off;
        off += (bytes + 255) & ~(size_t)255;
        return p;
    };
    unsigned short* wihf_bf = (unsigned short*)carve((size_t)G3 * E * 2);
    unsigned short* wihb_bf = (unsigned short*)carve((size_t)G3 * E * 2);
    unsigned short* whhf_bf = (unsigned short*)carve((size_t)G3 * H * 2);
    unsigned short* whhb_bf = (unsigned short*)carve((size_t)G3 * H * 2);
    float* xg_f = (float*)carve((size_t)MROWS * G3 * 4);
    float* xg_b = (float*)carve((size_t)MROWS * G3 * 4);
    float* hf   = (float*)carve((size_t)MROWS * H * 4);
    float* hb   = (float*)carve((size_t)MROWS * H * 4);
    float* emis = (float*)carve((size_t)MROWS * KTAG * 4);

    // 0) weight conversion to bf16 fragment-major
    conv_frag_kernel<<<(G3 * E + 255) / 256, 256, 0, stream>>>(w_ih_f, wihf_bf, G3, E);
    conv_frag_kernel<<<(G3 * E + 255) / 256, 256, 0, stream>>>(w_ih_b, wihb_bf, G3, E);
    conv_frag_kernel<<<(G3 * H + 255) / 256, 256, 0, stream>>>(w_hh_f, whhf_bf, G3, H);
    conv_frag_kernel<<<(G3 * H + 255) / 256, 256, 0, stream>>>(w_hh_b, whhb_bf, G3, H);

    // 1) embedding + input-gate GEMMs (WMMA)
    embed_xg_kernel<<<dim3(MROWS / 16, 2), 256, 0, stream>>>(
        x, etab, wihf_bf, wihb_bf, b_ih_f, b_ih_b, xg_f, xg_b);

    // 2) persistent bidirectional GRU (WMMA recurrence)
    gru_kernel<<<2, 1024, 0, stream>>>(xg_f, xg_b, whhf_bf, whhb_bf,
                                       b_hh_f, b_hh_b, hf, hb);

    // 3) linear head + softmax (async LDS staging of w_lin)
    linsoft_kernel<<<MROWS / 8, 256, 0, stream>>>(hf, hb, w_lin, b_lin, emis);

    // 4) CRF forward + loss
    crf_kernel<<<1, 64, 0, stream>>>(x, emis, trans, out);
}
